// CausalWanSelfAttention_34703335752398
// MI455X (gfx1250) — matrix operand order repaired and checked by reading, not compiled, not run on hardware
//
#include <hip/hip_runtime.h>
#include <hip/hip_bf16.h>
#include <cmath>

typedef __bf16 bf16_t;
typedef __attribute__((ext_vector_type(16))) __bf16 v16bf;
typedef __attribute__((ext_vector_type(8)))  float  v8f;

union BfOp { v16bf v; uint4 q[2]; };

__device__ __forceinline__ v8f wmma_bf16(v16bf a, v16bf b, v8f c) {
  // D = A(16x32) * B(32x16) + C, f32 accumulate
  return __builtin_amdgcn_wmma_f32_16x16x32_bf16(false, a, false, b, (short)0, c, false, false);
}

__device__ __forceinline__ void splitw(bf16_t* h, bf16_t* l, float x) {
  bf16_t hi = (bf16_t)x;
  *h = hi;
  *l = (bf16_t)(x - (float)hi);
}

// ---------------- elementwise f32 -> (hi,lo) bf16 split ----------------
__global__ void split_f32_kernel(const float* __restrict__ src,
                                 bf16_t* __restrict__ h, bf16_t* __restrict__ l,
                                 size_t n) {
  size_t i = (size_t)blockIdx.x * blockDim.x + threadIdx.x;
  if (i < n) splitw(h + i, l + i, src[i]);
}

// ---------------- LDS-tiled transpose + split: T[j][i] = W[i][j] ----------------
__global__ void __launch_bounds__(256) transpose_split_kernel(
    const float* __restrict__ W, bf16_t* __restrict__ Th, bf16_t* __restrict__ Tl, int C) {
  __shared__ float tile[32][33];
  int bx = blockIdx.x * 32, by = blockIdx.y * 32;
  int tx = threadIdx.x & 31, ty = threadIdx.x >> 5;   // 8 row-groups
  for (int r = ty; r < 32; r += 8)
    tile[r][tx] = W[(size_t)(by + r) * C + (bx + tx)];
  __syncthreads();
  for (int r = ty; r < 32; r += 8) {
    size_t o = (size_t)(bx + r) * C + (by + tx);
    splitw(Th + o, Tl + o, tile[tx][r]);
  }
}

// ---------------- split-bf16 GEMM: Y = A(MxK) * B^T(NoutxK) + bias ----------------
// one wave per 32x32 output block (2x2 WMMA tiles, register-blocked), 8 waves/block
__global__ void __launch_bounds__(256) gemm_bf16x3_kernel(
    const bf16_t* __restrict__ Ah, const bf16_t* __restrict__ Al,
    const bf16_t* __restrict__ Bh, const bf16_t* __restrict__ Bl,   // [Nout][K]
    const float* __restrict__ bias, float* __restrict__ Y,
    int M, int Nout, int K) {
  int wave = threadIdx.x >> 5;
  int lane = threadIdx.x & 31;
  int half = lane >> 4, lr = lane & 15;
  int m0 = blockIdx.x * 32;                  // row tiles m0, m0+16
  int n0 = (blockIdx.y * 8 + wave) * 32;     // col tiles n0, n0+16
  const bf16_t* ah0 = Ah + (size_t)(m0 + lr) * K;
  const bf16_t* al0 = Al + (size_t)(m0 + lr) * K;
  const bf16_t* ah1 = ah0 + (size_t)16 * K;
  const bf16_t* al1 = al0 + (size_t)16 * K;
  const bf16_t* bh0 = Bh + (size_t)(n0 + lr) * K;
  const bf16_t* bl0 = Bl + (size_t)(n0 + lr) * K;
  const bf16_t* bh1 = bh0 + (size_t)16 * K;
  const bf16_t* bl1 = bl0 + (size_t)16 * K;
  const v8f vzero = {0.f,0.f,0.f,0.f,0.f,0.f,0.f,0.f};
  v8f acc[2][2];
  acc[0][0] = vzero; acc[0][1] = vzero; acc[1][0] = vzero; acc[1][1] = vzero;
  for (int kc = 0; kc < K; kc += 32) {
    int ka = kc + half * 16;   // A frag: K = {ka..ka+15} contiguous
    int kb = kc + half * 16;   // B frag: K = {kb..kb+15} contiguous
    BfOp A0h, A0l, A1h, A1l, B0h, B0l, B1h, B1l;
    A0h.q[0] = *(const uint4*)(ah0 + ka); A0h.q[1] = *(const uint4*)(ah0 + ka + 8);
    A0l.q[0] = *(const uint4*)(al0 + ka); A0l.q[1] = *(const uint4*)(al0 + ka + 8);
    A1h.q[0] = *(const uint4*)(ah1 + ka); A1h.q[1] = *(const uint4*)(ah1 + ka + 8);
    A1l.q[0] = *(const uint4*)(al1 + ka); A1l.q[1] = *(const uint4*)(al1 + ka + 8);
    B0h.q[0] = *(const uint4*)(bh0 + kb); B0h.q[1] = *(const uint4*)(bh0 + kb + 8);
    B0l.q[0] = *(const uint4*)(bl0 + kb); B0l.q[1] = *(const uint4*)(bl0 + kb + 8);
    B1h.q[0] = *(const uint4*)(bh1 + kb); B1h.q[1] = *(const uint4*)(bh1 + kb + 8);
    B1l.q[0] = *(const uint4*)(bl1 + kb); B1l.q[1] = *(const uint4*)(bl1 + kb + 8);
    acc[0][0] = wmma_bf16(A0h.v, B0h.v, acc[0][0]);
    acc[0][0] = wmma_bf16(A0h.v, B0l.v, acc[0][0]);
    acc[0][0] = wmma_bf16(A0l.v, B0h.v, acc[0][0]);
    acc[0][1] = wmma_bf16(A0h.v, B1h.v, acc[0][1]);
    acc[0][1] = wmma_bf16(A0h.v, B1l.v, acc[0][1]);
    acc[0][1] = wmma_bf16(A0l.v, B1h.v, acc[0][1]);
    acc[1][0] = wmma_bf16(A1h.v, B0h.v, acc[1][0]);
    acc[1][0] = wmma_bf16(A1h.v, B0l.v, acc[1][0]);
    acc[1][0] = wmma_bf16(A1l.v, B0h.v, acc[1][0]);
    acc[1][1] = wmma_bf16(A1h.v, B1h.v, acc[1][1]);
    acc[1][1] = wmma_bf16(A1h.v, B1l.v, acc[1][1]);
    acc[1][1] = wmma_bf16(A1l.v, B1h.v, acc[1][1]);
  }
  for (int ni = 0; ni < 2; ++ni) {
    int bcol = n0 + ni * 16 + lr;
    float b = bias[bcol];
    for (int mi = 0; mi < 2; ++mi) {
      for (int r = 0; r < 8; ++r) {
        int orow = m0 + mi * 16 + r + 8 * half;   // C/D layout: VGPR r -> row r (+8 hi half)
        Y[(size_t)orow * Nout + bcol] = acc[mi][ni][r] + b;
      }
    }
  }
}

// ---------------- fill bf16 split KV from original caches ----------------
// cache layout [S][N][D] -> K: [N][S][D], V(transposed): [N][D][S]
__global__ void fill_cache_kernel(const float* __restrict__ ck, const float* __restrict__ cv,
                                  bf16_t* __restrict__ Kh, bf16_t* __restrict__ Kl,
                                  bf16_t* __restrict__ Vh, bf16_t* __restrict__ Vl,
                                  int S, int N, int D) {
  size_t i = (size_t)blockIdx.x * blockDim.x + threadIdx.x;
  size_t total = (size_t)S * N * D;
  if (i >= total) return;
  int d = (int)(i % D);
  size_t sn = i / D;
  int n = (int)(sn % N);
  int s = (int)(sn / N);
  size_t ko = ((size_t)n * S + s) * D + d;
  splitw(Kh + ko, Kl + ko, ck[i]);
  size_t vo = ((size_t)n * D + d) * S + s;
  splitw(Vh + vo, Vl + vo, cv[i]);
}

// ---------------- RMSNorm + gain + RoPE, scatter K/V into caches, emit Q ----------------
__global__ void __launch_bounds__(256) norm_rope_scatter_kernel(
    const float* __restrict__ qr, const float* __restrict__ kr, const float* __restrict__ vr,
    const float* __restrict__ gq, const float* __restrict__ gk,
    const float* __restrict__ theta, const int* __restrict__ widx,
    bf16_t* __restrict__ Qh, bf16_t* __restrict__ Ql,    // [N][L][D]
    bf16_t* __restrict__ Kh, bf16_t* __restrict__ Kl,    // [N][S][D]
    bf16_t* __restrict__ Vh, bf16_t* __restrict__ Vl,    // [N][D][S]
    int L, int C, int N, int D, int S) {
  int l = blockIdx.x;
  int t = threadIdx.x;
  __shared__ float red[256];
  const float* q = qr + (size_t)l * C;
  const float* k = kr + (size_t)l * C;
  float sq = 0.f, sk = 0.f;
  for (int c = t; c < C; c += 256) { float a = q[c]; sq += a * a; float b = k[c]; sk += b * b; }
  red[t] = sq; __syncthreads();
  for (int s = 128; s > 0; s >>= 1) { if (t < s) red[t] += red[t + s]; __syncthreads(); }
  float rq = rsqrtf(red[0] / (float)C + 1e-6f);
  __syncthreads();
  red[t] = sk; __syncthreads();
  for (int s = 128; s > 0; s >>= 1) { if (t < s) red[t] += red[t + s]; __syncthreads(); }
  float rk = rsqrtf(red[0] / (float)C + 1e-6f);

  int sidx = widx[l];
  int hD = D >> 1;
  for (int p = t; p < C / 2; p += 256) {
    int n = p / hD, i = p - n * hD;
    int c0 = 2 * p, c1 = c0 + 1;
    float th = theta[(size_t)l * hD + i];
    float ct = cosf(th), st = sinf(th);
    float q1 = q[c0] * rq * gq[c0], q2 = q[c1] * rq * gq[c1];
    float k1 = k[c0] * rk * gk[c0], k2 = k[c1] * rk * gk[c1];
    float qe = q1 * ct - q2 * st, qo = q1 * st + q2 * ct;
    float ke = k1 * ct - k2 * st, ko = k1 * st + k2 * ct;
    size_t qb = ((size_t)n * L + l) * D + 2 * i;
    splitw(Qh + qb,     Ql + qb,     qe);
    splitw(Qh + qb + 1, Ql + qb + 1, qo);
    size_t kb = ((size_t)n * S + sidx) * D + 2 * i;
    splitw(Kh + kb,     Kl + kb,     ke);
    splitw(Kh + kb + 1, Kl + kb + 1, ko);
  }
  for (int c = t; c < C; c += 256) {
    int n = c / D, d = c - n * D;
    size_t vb = ((size_t)n * D + d) * S + sidx;
    splitw(Vh + vb, Vl + vb, vr[(size_t)l * C + c]);
  }
}

// ---------------- flash attention ----------------
// 4 waves/block, same head, adjacent 16-row q-tiles; per-iteration barrier keeps
// all 4 waves on the same K/V chunk so 3/4 of K/V fetches hit the WGP cache.
__global__ void __launch_bounds__(128) flash_attn_kernel(
    const bf16_t* __restrict__ Qh, const bf16_t* __restrict__ Ql,   // [N][L][D]
    const bf16_t* __restrict__ Kh, const bf16_t* __restrict__ Kl,   // [N][S][D]
    const bf16_t* __restrict__ Vh, const bf16_t* __restrict__ Vl,   // [N][D][S]
    const unsigned char* __restrict__ mask,                         // [S]
    float* __restrict__ Out,                                        // [L][N*D]
    int L, int S, int N, int D, float scale) {
  int wave = threadIdx.x >> 5;
  int lane = threadIdx.x & 31;
  int n = blockIdx.y;
  int l0 = (blockIdx.x * 4 + wave) * 16;
  int half = lane >> 4, lr = lane & 15;

  __shared__ bf16_t PhiAll[4][16][32];
  __shared__ bf16_t PloAll[4][16][32];
  bf16_t (*Phi)[32] = PhiAll[wave];
  bf16_t (*Plo)[32] = PloAll[wave];

  // Q A-operands: 4 K-chunks of 32 (D=128), hi/lo split; reused over all of S
  BfOp qh[4], ql[4];
  const bf16_t* qrh = Qh + ((size_t)n * L + (l0 + lr)) * D;
  const bf16_t* qrl = Ql + ((size_t)n * L + (l0 + lr)) * D;
  int KC = D / 32;
  for (int kc = 0; kc < KC; ++kc) {
    int ka = kc * 32 + half * 16;
    qh[kc].q[0] = *(const uint4*)(qrh + ka); qh[kc].q[1] = *(const uint4*)(qrh + ka + 8);
    ql[kc].q[0] = *(const uint4*)(qrl + ka); ql[kc].q[1] = *(const uint4*)(qrl + ka + 8);
  }

  const v8f vzero = {0.f,0.f,0.f,0.f,0.f,0.f,0.f,0.f};
  v8f O[8];
  float m_r[8], l_r[8];
  int DT = D / 16;
  for (int t = 0; t < 8; ++t) O[t] = vzero;
  for (int r = 0; r < 8; ++r) { m_r[r] = -INFINITY; l_r[r] = 0.f; }

  const bf16_t* KhB = Kh + (size_t)n * S * D;
  const bf16_t* KlB = Kl + (size_t)n * S * D;
  const bf16_t* VhB = Vh + (size_t)n * D * S;
  const bf16_t* VlB = Vl + (size_t)n * D * S;

  for (int s0 = 0; s0 < S; s0 += 32) {
    if (s0 + 32 < S && wave == 0) {   // pull next chunk toward L2/L0 (global_prefetch_b8)
      __builtin_prefetch(KhB + (size_t)(s0 + 32 + lr) * D, 0, 0);
      __builtin_prefetch(VhB + (size_t)lr * S + s0 + 32, 0, 0);
    }
    // ---- scores: two 16x16 column tiles, bf16x3 over K=D ----
    float sc[2][8];
    for (int ct = 0; ct < 2; ++ct) {
      int col = s0 + ct * 16 + lr;
      const bf16_t* krh = KhB + (size_t)col * D;
      const bf16_t* krl = KlB + (size_t)col * D;
      v8f a = vzero;
      for (int kc = 0; kc < KC; ++kc) {
        int kb = kc * 32 + half * 16;
        BfOp Bhi, Blo;
        Bhi.q[0] = *(const uint4*)(krh + kb); Bhi.q[1] = *(const uint4*)(krh + kb + 8);
        Blo.q[0] = *(const uint4*)(krl + kb); Blo.q[1] = *(const uint4*)(krl + kb + 8);
        a = wmma_bf16(qh[kc].v, Bhi.v, a);
        a = wmma_bf16(qh[kc].v, Blo.v, a);
        a = wmma_bf16(ql[kc].v, Bhi.v, a);
      }
      float msk = mask[col] ? 0.f : -1e30f;
      for (int r = 0; r < 8; ++r) sc[ct][r] = a[r] * scale + msk;
    }
    // ---- online softmax (row = r + 8*half, reductions stay within 16-lane half) ----
    float corr[8];
    for (int r = 0; r < 8; ++r) {
      float mx = fmaxf(sc[0][r], sc[1][r]);
      mx = fmaxf(mx, __shfl_xor(mx, 1));
      mx = fmaxf(mx, __shfl_xor(mx, 2));
      mx = fmaxf(mx, __shfl_xor(mx, 4));
      mx = fmaxf(mx, __shfl_xor(mx, 8));
      float mnew = fmaxf(m_r[r], mx);
      corr[r] = __expf(m_r[r] - mnew);
      m_r[r] = mnew;
      float p0 = __expf(sc[0][r] - mnew);
      float p1 = __expf(sc[1][r] - mnew);
      float sum = p0 + p1;
      sum += __shfl_xor(sum, 1);
      sum += __shfl_xor(sum, 2);
      sum += __shfl_xor(sum, 4);
      sum += __shfl_xor(sum, 8);
      l_r[r] = l_r[r] * corr[r] + sum;
      int row = r + 8 * half;
      splitw(&Phi[row][lr],      &Plo[row][lr],      p0);
      splitw(&Phi[row][16 + lr], &Plo[row][16 + lr], p1);
    }
    for (int t = 0; t < DT; ++t)
      for (int r = 0; r < 8; ++r) O[t][r] *= corr[r];
    __syncthreads();
    // ---- P back out of LDS in A-operand layout (K=32) ----
    BfOp Pha, Pla;
    Pha.q[0] = *(const uint4*)(&Phi[lr][half * 16]);
    Pha.q[1] = *(const uint4*)(&Phi[lr][half * 16 + 8]);
    Pla.q[0] = *(const uint4*)(&Plo[lr][half * 16]);
    Pla.q[1] = *(const uint4*)(&Plo[lr][half * 16 + 8]);
    // ---- O += P * V over this 32-wide S chunk ----
    for (int t = 0; t < DT; ++t) {
      int d = t * 16 + lr;
      const bf16_t* vrh = VhB + (size_t)d * S + s0 + half * 16;
      const bf16_t* vrl = VlB + (size_t)d * S + s0 + half * 16;
      BfOp Bvh, Bvl;
      Bvh.q[0] = *(const uint4*)(vrh); Bvh.q[1] = *(const uint4*)(vrh + 8);
      Bvl.q[0] = *(const uint4*)(vrl); Bvl.q[1] = *(const uint4*)(vrl + 8);
      O[t] = wmma_bf16(Pha.v, Bvh.v, O[t]);
      O[t] = wmma_bf16(Pha.v, Bvl.v, O[t]);
      O[t] = wmma_bf16(Pla.v, Bvh.v, O[t]);
    }
    __syncthreads();
  }
  for (int r = 0; r < 8; ++r) {
    float inv = (l_r[r] > 0.f) ? 1.f / l_r[r] : 0.f;
    int orow = l0 + r + 8 * half;
    for (int t = 0; t < DT; ++t)
      Out[(size_t)orow * ((size_t)N * D) + (size_t)n * D + t * 16 + lr] = O[t][r] * inv;
  }
}

// =====================================================================
extern "C" void kernel_launch(void* const* d_in, const int* in_sizes, int n_in,
                              void* d_out, int out_size, void* d_ws, size_t ws_size,
                              hipStream_t stream) {
  const float* x        = (const float*)d_in[0];
  const float* cache_k  = (const float*)d_in[1];
  const float* cache_v  = (const float*)d_in[2];
  const int*   widx     = (const int*)d_in[3];
  const unsigned char* mask = (const unsigned char*)d_in[4];
  const float* theta    = (const float*)d_in[5];
  const float* Wq = (const float*)d_in[6];
  const float* bq = (const float*)d_in[7];
  const float* Wk = (const float*)d_in[8];
  const float* bk = (const float*)d_in[9];
  const float* Wv = (const float*)d_in[10];
  const float* bv = (const float*)d_in[11];
  const float* Wo = (const float*)d_in[12];
  const float* bo = (const float*)d_in[13];
  const float* gq = (const float*)d_in[14];
  const float* gk = (const float*)d_in[15];

  int L = in_sizes[3];                 // write_indices: (L,)
  int S = in_sizes[4];                 // attn_mask: (1,1,1,S)
  int C = in_sizes[0] / L;             // x: (1,L,C)
  int D = 2 * (in_sizes[5] / L);       // rope_theta: (L,1,D/2)
  int N = C / D;

  char* p = (char*)d_ws;
  auto alloc = [&](size_t bytes) { void* r = (void*)p; p += (bytes + 255) & ~(size_t)255; return r; };
  size_t LC  = (size_t)L * C;
  size_t CC  = (size_t)C * C;
  size_t NSD = (size_t)N * S * D;

  bf16_t* Xh = (bf16_t*)alloc(LC * 2);  bf16_t* Xl = (bf16_t*)alloc(LC * 2);
  bf16_t* Wt[8];
  for (int i = 0; i < 8; ++i) Wt[i] = (bf16_t*)alloc(CC * 2);   // q,k,v,o x (hi,lo), transposed
  float* qraw = (float*)alloc(LC * 4);
  float* kraw = (float*)alloc(LC * 4);
  float* vraw = (float*)alloc(LC * 4);
  bf16_t* Qhh = (bf16_t*)alloc(LC * 2);  bf16_t* Qll = (bf16_t*)alloc(LC * 2);
  bf16_t* Khh = (bf16_t*)alloc(NSD * 2); bf16_t* Kll = (bf16_t*)alloc(NSD * 2);
  bf16_t* Vhh = (bf16_t*)alloc(NSD * 2); bf16_t* Vll = (bf16_t*)alloc(NSD * 2);
  float*  AO  = (float*)alloc(LC * 4);
  bf16_t* AOh = (bf16_t*)alloc(LC * 2);  bf16_t* AOl = (bf16_t*)alloc(LC * 2);

  // 1) split activations
  split_f32_kernel<<<(int)((LC + 255) / 256), 256, 0, stream>>>(x, Xh, Xl, LC);
  // 2) transpose+split weights
  {
    dim3 g(C / 32, C / 32);
    transpose_split_kernel<<<g, 256, 0, stream>>>(Wq, Wt[0], Wt[1], C);
    transpose_split_kernel<<<g, 256, 0, stream>>>(Wk, Wt[2], Wt[3], C);
    transpose_split_kernel<<<g, 256, 0, stream>>>(Wv, Wt[4], Wt[5], C);
    transpose_split_kernel<<<g, 256, 0, stream>>>(Wo, Wt[6], Wt[7], C);
  }
  // 3) QKV projections (WMMA, bf16x3, 2x2 register-blocked)
  {
    dim3 g(L / 32, C / 256);
    gemm_bf16x3_kernel<<<g, 256, 0, stream>>>(Xh, Xl, Wt[0], Wt[1], bq, qraw, L, C, C);
    gemm_bf16x3_kernel<<<g, 256, 0, stream>>>(Xh, Xl, Wt[2], Wt[3], bk, kraw, L, C, C);
    gemm_bf16x3_kernel<<<g, 256, 0, stream>>>(Xh, Xl, Wt[4], Wt[5], bv, vraw, L, C, C);
  }
  // 4) convert original caches to split-bf16 (K row-major, V transposed)
  fill_cache_kernel<<<(int)((NSD + 255) / 256), 256, 0, stream>>>(
      cache_k, cache_v, Khh, Kll, Vhh, Vll, S, N, D);
  // 5) RMSNorm + RoPE, scatter new K/V rows at write_indices, emit Q
  norm_rope_scatter_kernel<<<L, 256, 0, stream>>>(
      qraw, kraw, vraw, gq, gk, theta, widx,
      Qhh, Qll, Khh, Kll, Vhh, Vll, L, C, N, D, S);
  // 6) flash attention (WMMA scores + WMMA PV), 4 q-tiles of one head per WGP
  {
    dim3 g(L / 64, N);
    float scale = 1.f / sqrtf((float)D);
    flash_attn_kernel<<<g, 128, 0, stream>>>(
        Qhh, Qll, Khh, Kll, Vhh, Vll, mask, AO, L, S, N, D, scale);
  }
  // 7) output projection
  split_f32_kernel<<<(int)((LC + 255) / 256), 256, 0, stream>>>(AO, AOh, AOl, LC);
  {
    dim3 g(L / 32, C / 256);
    gemm_bf16x3_kernel<<<g, 256, 0, stream>>>(AOh, AOl, Wt[6], Wt[7], bo, (float*)d_out, L, C, C);
  }
}
